// BaseSelfAttention_62878321214088
// MI455X (gfx1250) — matrix operand order repaired and checked by reading, not compiled, not run on hardware
//
#include <hip/hip_runtime.h>
#include <hip/hip_bf16.h>

typedef __bf16 bf16_t;
typedef bf16_t v16bf __attribute__((ext_vector_type(16)));
typedef float  v8f   __attribute__((ext_vector_type(8)));

#define B_  8
#define T_  1024
#define D_  1024
#define H_  16
#define HD_ 64
#define TE_ 2048
#define M_  (B_*T_)
#define LN_EPS 1e-5f

// ---------------- WMMA helpers ----------------

__device__ inline v8f wmma_bf16(v16bf a, v16bf b, v8f c) {
  return __builtin_amdgcn_wmma_f32_16x16x32_bf16(
      /*neg_a=*/false, a, /*neg_b=*/false, b,
      /*c_mod=*/(short)0, c, /*reuse_a=*/false, /*reuse_b=*/false);
}

// A fragment: 16x32 bf16 tile, row-major source, p = tile origin, stride lda.
// lane m = lane&15 ; K per lane: 16*(lane>>4) + j, j=0..15 contiguous
__device__ inline v16bf frag_a(const bf16_t* p, int lda, int lane) {
  const bf16_t* row = p + (size_t)(lane & 15) * lda + ((lane >> 4) << 4);
  v16bf f;
#pragma unroll
  for (int j = 0; j < 16; ++j) f[j] = row[j];
  return f;
}

// B fragment 32x16 from B^T storage (n-major, contiguous K): p = tile origin (n0,k0)
// lane n = lane&15 ; K = 16*(lane>>4) + j, j=0..15 contiguous
__device__ inline v16bf frag_bT(const bf16_t* p, int ldt, int lane) {
  const bf16_t* row = p + (size_t)(lane & 15) * ldt + ((lane >> 4) << 4);
  v16bf f;
#pragma unroll
  for (int j = 0; j < 16; ++j) f[j] = row[j];
  return f;
}

// ---------------- Kernel 1: LayerNorm(x) -> bf16 ----------------

__global__ void ln_bf16_kernel(const float* __restrict__ x,
                               const float* __restrict__ w,
                               const float* __restrict__ b,
                               bf16_t* __restrict__ out) {
  int row = blockIdx.x;
  const float* xr = x + (size_t)row * D_;
  float s = 0.f, ss = 0.f;
  for (int i = threadIdx.x; i < D_; i += 256) { float v = xr[i]; s += v; ss += v * v; }
  __shared__ float rs[256], rss[256];
  rs[threadIdx.x] = s; rss[threadIdx.x] = ss;
  __syncthreads();
  for (int st = 128; st > 0; st >>= 1) {
    if (threadIdx.x < st) { rs[threadIdx.x] += rs[threadIdx.x + st]; rss[threadIdx.x] += rss[threadIdx.x + st]; }
    __syncthreads();
  }
  float mean = rs[0] * (1.f / D_);
  float var  = rss[0] * (1.f / D_) - mean * mean;
  float rstd = rsqrtf(var + LN_EPS);
  for (int i = threadIdx.x; i < D_; i += 256)
    out[(size_t)row * D_ + i] = (bf16_t)((xr[i] - mean) * rstd * w[i] + b[i]);
}

// ---------------- Kernel 2: weight transpose + fp32->bf16 ----------------

__global__ void transpose_bf16_kernel(const float* __restrict__ W,
                                      bf16_t* __restrict__ Wt, int K, int N) {
  __shared__ float t[32][33];
  int k0 = blockIdx.x * 32, n0 = blockIdx.y * 32;
  int tx = threadIdx.x & 31, ty = threadIdx.x >> 5;   // 32x8 threads
  for (int i = ty; i < 32; i += 8) t[i][tx] = W[(size_t)(k0 + i) * N + n0 + tx];
  __syncthreads();
  for (int i = ty; i < 32; i += 8) Wt[(size_t)(n0 + i) * K + k0 + tx] = (bf16_t)t[tx][i];
}

// ---------------- Kernel 2b: v [B,T,H,64] -> vt [B,H,64,T] (bf16) ----------------

__global__ void vtranspose_kernel(const bf16_t* __restrict__ v, bf16_t* __restrict__ vt) {
  __shared__ bf16_t t[32][33];
  int bh = blockIdx.z, b = bh >> 4, h = bh & (H_ - 1);
  int m0 = blockIdx.x * 32, l0 = blockIdx.y * 32;
  int tx = threadIdx.x & 31, ty = threadIdx.x >> 5;
  for (int i = ty; i < 32; i += 8)
    t[i][tx] = v[((size_t)(b * T_ + m0 + i) * H_ + h) * HD_ + l0 + tx];
  __syncthreads();
  for (int i = ty; i < 32; i += 8)
    vt[((size_t)bh * HD_ + l0 + i) * T_ + m0 + tx] = t[tx][i];
}

// ---------------- Kernel 3/7: WMMA GEMM  C[M,N] = A[M,K] @ Bt^T + bias ----------------
// Block: 256 threads (8 waves), tile 256(M) x 64(N); wave w -> rows w*32..w*32+31.
// Software-pipelined, double-buffered LDS; __launch_bounds__(256,1) trades occupancy
// for registers so the pipeline (2 acc banks + staging temps) stays spill-free.
// outBf != nullptr : bf16 output (optionally row-scaled by mask)
// outBf == nullptr : fp32 output = resid + gemm + bias

__global__ void __launch_bounds__(256, 1)
gemm_wmma_kernel(const bf16_t* __restrict__ A,
                 const bf16_t* __restrict__ Bt,
                 const float* __restrict__ bias,
                 bf16_t* __restrict__ outBf,
                 float* __restrict__ outF,
                 const float* __restrict__ mask,
                 const float* __restrict__ resid,
                 int K, int N) {
  __shared__ __align__(16) bf16_t lA[2][256 * 40];
  __shared__ __align__(16) bf16_t lB[2][64 * 40];
  int tid = threadIdx.x, lane = tid & 31, wave = tid >> 5;
  int bm = blockIdx.x * 256, bn = blockIdx.y * 64;
  const bf16_t* aRow = A + (size_t)(bm + tid) * K;                       // one 32-col row per thread
  const bf16_t* bRow = Bt + (size_t)(bn + (tid >> 2)) * K + (tid & 3) * 8;
  const int NT = K / 32;

  // prologue: stage tile 0
  uint4 ra0, ra1, ra2, ra3, rb;
  {
    const uint4* s = reinterpret_cast<const uint4*>(aRow);
    ra0 = s[0]; ra1 = s[1]; ra2 = s[2]; ra3 = s[3];
    rb = *reinterpret_cast<const uint4*>(bRow);
    uint4* d = reinterpret_cast<uint4*>(&lA[0][tid * 40]);
    d[0] = ra0; d[1] = ra1; d[2] = ra2; d[3] = ra3;
    *reinterpret_cast<uint4*>(&lB[0][(tid >> 2) * 40 + (tid & 3) * 8]) = rb;
  }
  __syncthreads();

  v8f acc[2][4] = {};
  for (int i = 0; i < NT; ++i) {
    int cur = i & 1, nxt = cur ^ 1;
    if (i + 1 < NT) {  // prefetch next tile into registers (latency hidden by WMMAs)
      const uint4* s = reinterpret_cast<const uint4*>(aRow + (i + 1) * 32);
      ra0 = s[0]; ra1 = s[1]; ra2 = s[2]; ra3 = s[3];
      rb = *reinterpret_cast<const uint4*>(bRow + (i + 1) * 32);
    }
    v16bf a0 = frag_a(&lA[cur][(wave * 32) * 40], 40, lane);
    v16bf a1 = frag_a(&lA[cur][(wave * 32 + 16) * 40], 40, lane);
    v16bf b0 = frag_bT(&lB[cur][0 * 40], 40, lane);
    v16bf b1 = frag_bT(&lB[cur][16 * 40], 40, lane);
    v16bf b2 = frag_bT(&lB[cur][32 * 40], 40, lane);
    v16bf b3 = frag_bT(&lB[cur][48 * 40], 40, lane);
    acc[0][0] = wmma_bf16(a0, b0, acc[0][0]);
    acc[0][1] = wmma_bf16(a0, b1, acc[0][1]);
    acc[0][2] = wmma_bf16(a0, b2, acc[0][2]);
    acc[0][3] = wmma_bf16(a0, b3, acc[0][3]);
    acc[1][0] = wmma_bf16(a1, b0, acc[1][0]);
    acc[1][1] = wmma_bf16(a1, b1, acc[1][1]);
    acc[1][2] = wmma_bf16(a1, b2, acc[1][2]);
    acc[1][3] = wmma_bf16(a1, b3, acc[1][3]);
    if (i + 1 < NT) {  // commit prefetched tile to the other LDS buffer
      uint4* d = reinterpret_cast<uint4*>(&lA[nxt][tid * 40]);
      d[0] = ra0; d[1] = ra1; d[2] = ra2; d[3] = ra3;
      *reinterpret_cast<uint4*>(&lB[nxt][(tid >> 2) * 40 + (tid & 3) * 8]) = rb;
    }
    __syncthreads();
  }

  int colBase = bn + (lane & 15);
#pragma unroll
  for (int i = 0; i < 2; ++i) {
    int rowBase = bm + wave * 32 + i * 16 + ((lane >> 4) << 3);
#pragma unroll
    for (int t = 0; t < 4; ++t) {
      int col = colBase + t * 16;
      float bv = bias[col];
#pragma unroll
      for (int r = 0; r < 8; ++r) {
        int row = rowBase + r;
        float val = acc[i][t][r] + bv;
        if (outBf) {
          if (mask) val *= mask[row];
          outBf[(size_t)row * N + col] = (bf16_t)val;
        } else {
          outF[(size_t)row * N + col] = resid[(size_t)row * N + col] + val;
        }
      }
    }
  }
}

// ---------------- Kernel 4: fused attention (flash-style, per-wave 16 queries) ----------------
// grid: (T/128, B*H), block 256 (8 waves). q/k: bf16 [B,T,H,64]; vt: bf16 [B,H,64,T]; y: f32 [B,T,D]

__global__ void __launch_bounds__(256, 1)
attention_kernel(const bf16_t* __restrict__ q,
                 const bf16_t* __restrict__ k,
                 const bf16_t* __restrict__ vt,
                 const float* __restrict__ mask,
                 float* __restrict__ y) {
  __shared__ __align__(16) bf16_t plds[8 * 16 * 40];  // per-wave 16x32 P tile (stride 40)
  int lane = threadIdx.x & 31, wave = threadIdx.x >> 5;
  int bh = blockIdx.y, b = bh >> 4, h = bh & (H_ - 1);
  int qm0 = blockIdx.x * 128 + wave * 16;
  const size_t ld = (size_t)H_ * HD_;  // 1024
  const bf16_t* qp  = q + ((size_t)b * T_ * H_ + h) * HD_;
  const bf16_t* kp  = k + ((size_t)b * T_ * H_ + h) * HD_;
  const bf16_t* vtp = vt + (size_t)bh * HD_ * T_;     // [64][T]
  bf16_t* P = &plds[wave * 16 * 40];

  v16bf qa0 = frag_a(qp + (size_t)qm0 * ld, (int)ld, lane);       // l = 0..31
  v16bf qa1 = frag_a(qp + (size_t)qm0 * ld + 32, (int)ld, lane);  // l = 32..63

  v8f o[4] = {};
  float mrow[8], srow[8];
#pragma unroll
  for (int r = 0; r < 8; ++r) { mrow[r] = -1e30f; srow[r] = 0.f; }

  for (int km = 0; km < T_; km += 32) {
    // S = Q @ K^T : two 16x16 key tiles; preload all 4 K fragments
    v16bf kb00 = frag_bT(kp + (size_t)km * ld, (int)ld, lane);
    v16bf kb01 = frag_bT(kp + (size_t)km * ld + 32, (int)ld, lane);
    v16bf kb10 = frag_bT(kp + (size_t)(km + 16) * ld, (int)ld, lane);
    v16bf kb11 = frag_bT(kp + (size_t)(km + 16) * ld + 32, (int)ld, lane);
    v8f s0 = {}, s1 = {};
    s0 = wmma_bf16(qa0, kb00, s0);
    s0 = wmma_bf16(qa1, kb01, s0);
    s1 = wmma_bf16(qa0, kb10, s1);
    s1 = wmma_bf16(qa1, kb11, s1);

    float add0 = (1.f - mask[b * T_ + km + (lane & 15)]) * -1000000.f;
    float add1 = (1.f - mask[b * T_ + km + 16 + (lane & 15)]) * -1000000.f;

#pragma unroll
    for (int r = 0; r < 8; ++r) {
      float a0 = s0[r] + add0, a1 = s1[r] + add1;
      float mx = fmaxf(a0, a1);
#pragma unroll
      for (int off = 1; off < 16; off <<= 1) mx = fmaxf(mx, __shfl_xor(mx, off, 32));
      float nm = fmaxf(mrow[r], mx);
      float alpha = __expf(mrow[r] - nm);
      float p0 = __expf(a0 - nm), p1 = __expf(a1 - nm);
      float ps = p0 + p1;
#pragma unroll
      for (int off = 1; off < 16; off <<= 1) ps += __shfl_xor(ps, off, 32);
      srow[r] = srow[r] * alpha + ps;
      mrow[r] = nm;
      o[0][r] *= alpha; o[1][r] *= alpha; o[2][r] *= alpha; o[3][r] *= alpha;
      int row = r + ((lane >> 4) << 3);
      P[row * 40 + (lane & 15)]      = (bf16_t)p0;
      P[row * 40 + 16 + (lane & 15)] = (bf16_t)p1;
    }
    asm volatile("s_wait_dscnt 0" ::: "memory");  // LDS RAW across lanes within wave
    v16bf pa = frag_a(P, 40, lane);
    // O += P @ V : V^T storage -> contiguous-K fragments
    v16bf vb0 = frag_bT(vtp + (size_t)0  * T_ + km, T_, lane);
    v16bf vb1 = frag_bT(vtp + (size_t)16 * T_ + km, T_, lane);
    v16bf vb2 = frag_bT(vtp + (size_t)32 * T_ + km, T_, lane);
    v16bf vb3 = frag_bT(vtp + (size_t)48 * T_ + km, T_, lane);
    o[0] = wmma_bf16(pa, vb0, o[0]);
    o[1] = wmma_bf16(pa, vb1, o[1]);
    o[2] = wmma_bf16(pa, vb2, o[2]);
    o[3] = wmma_bf16(pa, vb3, o[3]);
  }

#pragma unroll
  for (int r = 0; r < 8; ++r) {
    int row = qm0 + r + ((lane >> 4) << 3);
    float inv = srow[r] > 0.f ? 1.f / srow[r] : 0.f;
    size_t base = ((size_t)b * T_ + row) * D_ + h * HD_;
#pragma unroll
    for (int t = 0; t < 4; ++t)
      y[base + t * 16 + (lane & 15)] = o[t][r] * inv;
  }
}

// ---------------- Kernel 5: e = silu(emb) @ emb_w + emb_b ----------------
// grid (2048/256, B), block 256

__global__ void emb_mlp_kernel(const float* __restrict__ emb,
                               const float* __restrict__ ew,
                               const float* __restrict__ eb,
                               float* __restrict__ e) {
  __shared__ float se[TE_];
  int b = blockIdx.y;
  for (int i = threadIdx.x; i < TE_; i += 256) {
    float v = emb[(size_t)b * TE_ + i];
    se[i] = v / (1.f + __expf(-v));
  }
  __syncthreads();
  int col = blockIdx.x * 256 + threadIdx.x;  // 0..2047 (2D)
  float acc = eb[col];
  for (int t = 0; t < TE_; ++t) acc += se[t] * ew[(size_t)t * (2 * D_) + col];
  e[(size_t)b * (2 * D_) + col] = acc;
}

// ---------------- Kernel 6: LN(y)*(1+scale)+shift -> silu -> bf16 ----------------

__global__ void ln_style_kernel(const float* __restrict__ y,
                                const float* __restrict__ w,
                                const float* __restrict__ b,
                                const float* __restrict__ e,
                                bf16_t* __restrict__ out) {
  int row = blockIdx.x;
  int bi = row >> 10;  // row / T_
  const float* yr = y + (size_t)row * D_;
  float s = 0.f, ss = 0.f;
  for (int i = threadIdx.x; i < D_; i += 256) { float v = yr[i]; s += v; ss += v * v; }
  __shared__ float rs[256], rss[256];
  rs[threadIdx.x] = s; rss[threadIdx.x] = ss;
  __syncthreads();
  for (int st = 128; st > 0; st >>= 1) {
    if (threadIdx.x < st) { rs[threadIdx.x] += rs[threadIdx.x + st]; rss[threadIdx.x] += rss[threadIdx.x + st]; }
    __syncthreads();
  }
  float mean = rs[0] * (1.f / D_);
  float var  = rss[0] * (1.f / D_) - mean * mean;
  float rstd = rsqrtf(var + LN_EPS);
  for (int i = threadIdx.x; i < D_; i += 256) {
    float yn = (yr[i] - mean) * rstd * w[i] + b[i];
    float sc = e[(size_t)bi * (2 * D_) + i];
    float sh = e[(size_t)bi * (2 * D_) + D_ + i];
    float hv = yn * (1.f + sc) + sh;
    out[(size_t)row * D_ + i] = (bf16_t)(hv / (1.f + __expf(-hv)));  // silu
  }
}

// ---------------- launch ----------------

extern "C" void kernel_launch(void* const* d_in, const int* in_sizes, int n_in,
                              void* d_out, int out_size, void* d_ws, size_t ws_size,
                              hipStream_t stream) {
  const float* x       = (const float*)d_in[0];
  const float* emb     = (const float*)d_in[1];
  const float* srcmask = (const float*)d_in[2];
  const float* norm_w  = (const float*)d_in[3];
  const float* norm_b  = (const float*)d_in[4];
  const float* q_w     = (const float*)d_in[5];
  const float* q_b     = (const float*)d_in[6];
  const float* k_w     = (const float*)d_in[7];
  const float* k_b     = (const float*)d_in[8];
  const float* v_w     = (const float*)d_in[9];
  const float* v_b     = (const float*)d_in[10];
  const float* emb_w   = (const float*)d_in[11];
  const float* emb_b   = (const float*)d_in[12];
  const float* sb_w    = (const float*)d_in[13];
  const float* sb_b    = (const float*)d_in[14];
  const float* out_w   = (const float*)d_in[15];
  const float* out_b   = (const float*)d_in[16];
  float* out           = (float*)d_out;

  // workspace layout
  char* w = (char*)d_ws;
  size_t off = 0;
  bf16_t* xn  = (bf16_t*)(w + off); off += (size_t)M_ * D_ * 2;
  bf16_t* qwT = (bf16_t*)(w + off); off += (size_t)D_ * D_ * 2;
  bf16_t* kwT = (bf16_t*)(w + off); off += (size_t)D_ * D_ * 2;
  bf16_t* vwT = (bf16_t*)(w + off); off += (size_t)D_ * D_ * 2;
  bf16_t* owT = (bf16_t*)(w + off); off += (size_t)D_ * D_ * 2;
  bf16_t* qb  = (bf16_t*)(w + off); off += (size_t)M_ * D_ * 2;
  bf16_t* kb  = (bf16_t*)(w + off); off += (size_t)M_ * D_ * 2;
  bf16_t* vb  = (bf16_t*)(w + off); off += (size_t)M_ * D_ * 2;
  bf16_t* vtb = (bf16_t*)(w + off); off += (size_t)M_ * D_ * 2;
  float*  yf  = (float*)(w + off);  off += (size_t)M_ * D_ * 4;
  float*  ef  = (float*)(w + off);  off += (size_t)B_ * 2 * D_ * 4;
  bf16_t* hb  = (bf16_t*)(w + off); off += (size_t)M_ * D_ * 2;
  (void)ws_size; (void)in_sizes; (void)n_in; (void)out_size;

  // 1) LayerNorm(x) -> bf16
  ln_bf16_kernel<<<M_, 256, 0, stream>>>(x, norm_w, norm_b, xn);

  // 2) weight transpose + convert
  dim3 tg(D_ / 32, D_ / 32);
  transpose_bf16_kernel<<<tg, 256, 0, stream>>>(q_w, qwT, D_, D_);
  transpose_bf16_kernel<<<tg, 256, 0, stream>>>(k_w, kwT, D_, D_);
  transpose_bf16_kernel<<<tg, 256, 0, stream>>>(v_w, vwT, D_, D_);
  transpose_bf16_kernel<<<tg, 256, 0, stream>>>(out_w, owT, D_, D_);

  // 3) Q/K/V projections (WMMA)
  dim3 gg(M_ / 256, D_ / 64);
  gemm_wmma_kernel<<<gg, 256, 0, stream>>>(xn, qwT, q_b, qb, nullptr, nullptr, nullptr, D_, D_);
  gemm_wmma_kernel<<<gg, 256, 0, stream>>>(xn, kwT, k_b, kb, nullptr, nullptr, nullptr, D_, D_);
  gemm_wmma_kernel<<<gg, 256, 0, stream>>>(xn, vwT, v_b, vb, nullptr, srcmask, nullptr, D_, D_);

  // 3b) transpose V for contiguous PV fragments
  dim3 gv(T_ / 32, HD_ / 32, B_ * H_);
  vtranspose_kernel<<<gv, 256, 0, stream>>>(vb, vtb);

  // 4) fused attention -> y (f32)
  dim3 ga(T_ / 128, B_ * H_);
  attention_kernel<<<ga, 256, 0, stream>>>(qb, kb, vtb, srcmask, yf);

  // 5) emb MLP -> e (f32 [B, 2D])
  dim3 ge(2 * D_ / 256, B_);
  emb_mlp_kernel<<<ge, 256, 0, stream>>>(emb, emb_w, emb_b, ef);

  // 6) stylization LN + silu -> bf16
  ln_style_kernel<<<M_, 256, 0, stream>>>(yf, sb_w, sb_b, ef, hb);

  // 7) output projection + residual (WMMA, f32 out)
  gemm_wmma_kernel<<<gg, 256, 0, stream>>>(hb, owT, out_b, nullptr, out, nullptr, x, D_, D_);
}
